// PointGNNLayer_23983097381128
// MI455X (gfx1250) — matrix operand order repaired and checked by reading, not compiled, not run on hardware
//
#include <hip/hip_runtime.h>
#include <hip/hip_bf16.h>

#define N_PTS 16384
#define KNN   16
#define PTS_PER_BLK 16

typedef __bf16 bf16_t;
typedef bf16_t v16bf __attribute__((ext_vector_type(16)));
typedef float  v8f   __attribute__((ext_vector_type(8)));

union BFrag { v16bf v; unsigned int u[8]; };

__device__ __forceinline__ unsigned short f2bf(float f) {
  unsigned int u = __float_as_uint(f);
  unsigned int r = u + 0x7FFFu + ((u >> 16) & 1u);   // round-to-nearest-even
  return (unsigned short)(r >> 16);
}

// ---------------- prep: f32 -> bf16 feature copy (vectorized) ----------------
__global__ void k_conv_x(const float4* __restrict__ x4, uint2* __restrict__ xb2, int n4) {
  int t = blockIdx.x * blockDim.x + threadIdx.x;
  if (t < n4) {
    float4 f = x4[t];
    uint2 o;
    o.x = (unsigned int)f2bf(f.x) | ((unsigned int)f2bf(f.y) << 16);
    o.y = (unsigned int)f2bf(f.z) | ((unsigned int)f2bf(f.w) << 16);
    xb2[t] = o;
  }
}

// ---------------- prep: swizzle weights into WMMA B-fragment order ----------------
// One block per (nt,kt) tile; dword d = lane*8+v holds bf16 pair (K0,K0+1) for column nt*16+(lane&15),
// K0 = kt*32 + 16*(v>>2) + (lane>=16 ? 8 : 0) + 2*(v&3).  Rows >= Krows are zero padding.
__global__ void k_pack_w(const float* __restrict__ W, unsigned int* __restrict__ dst,
                         int Krows, int Ktiles) {
  int t   = blockIdx.x;
  int kt  = t % Ktiles, nt = t / Ktiles;
  int tid = threadIdx.x;          // 0..255
  int lane = tid >> 3, v = tid & 7;
  int half = lane >> 4;
  int n  = nt * 16 + (lane & 15);
  int K0 = kt * 32 + (v >> 2) * 16 + half * 8 + 2 * (v & 3);
  unsigned int lo = (K0     < Krows) ? (unsigned int)f2bf(W[(size_t)K0 * 128 + n])       : 0u;
  unsigned int hi = (K0 + 1 < Krows) ? (unsigned int)f2bf(W[(size_t)(K0 + 1) * 128 + n]) : 0u;
  dst[t * 256 + tid] = lo | (hi << 16);
}

// ---------------- kNN: one wave per row, per-lane sorted top-16, shfl merge ----------------
__global__ void __launch_bounds__(256) k_knn(const float* __restrict__ coords,
                                             int* __restrict__ knn) {
  const int lane = threadIdx.x & 31;
  const int wv   = threadIdx.x >> 5;
  const int i    = blockIdx.x * 8 + wv;
  const float4 pc = *(const float4*)(coords + (size_t)i * 4);
  const float sqi = pc.x * pc.x + pc.y * pc.y + pc.z * pc.z;

  unsigned long long best[16];
#pragma unroll
  for (int t = 0; t < 16; ++t) best[t] = ~0ull;

  for (int j = lane; j < N_PTS; j += 32) {
    float4 qc = *(const float4*)(coords + (size_t)j * 4);
    float sqj = qc.x * qc.x + qc.y * qc.y + qc.z * qc.z;
    float dot = pc.x * qc.x + pc.y * qc.y + pc.z * qc.z;
    float d2  = sqi + sqj - 2.0f * dot;
    unsigned int b = __float_as_uint(d2);
    b ^= ((unsigned int)((int)b >> 31)) | 0x80000000u;      // order-preserving float->uint
    unsigned long long key = ((unsigned long long)b << 14) | (unsigned int)j;
    if (key < best[15]) {                                    // single bubble pass keeps list sorted
      best[15] = key;
#pragma unroll
      for (int t = 15; t > 0; --t) {
        unsigned long long a = best[t - 1], c = best[t];
        best[t - 1] = a < c ? a : c;
        best[t]     = a < c ? c : a;
      }
    }
  }
  // 32-way merge of per-lane sorted lists: 16 rounds of cross-lane min
#pragma unroll 1
  for (int t = 0; t < 16; ++t) {
    unsigned long long k = best[0];
    unsigned long long m = k;
#pragma unroll
    for (int s = 16; s >= 1; s >>= 1) {
      unsigned long long o = __shfl_xor(m, s, 32);
      m = o < m ? o : m;
    }
    if (k == m) {                       // unique winner (idx bits make keys distinct)
#pragma unroll
      for (int q = 0; q < 15; ++q) best[q] = best[q + 1];
      best[15] = ~0ull;
    }
    if (lane == 0) knn[i * KNN + t] = (int)(m & 0x3FFFull);
  }
}

// ---------------- hoisted self-term: selfc = x @ mW1[0:128]  (N x 128, f32) ----------------
__global__ void __launch_bounds__(256) k_selfc(const unsigned short* __restrict__ xb,
                                               const unsigned int* __restrict__ w1ap,
                                               float* __restrict__ selfc) {
  __shared__ __align__(16) unsigned short abuf[16 * 128];
  const int tid  = threadIdx.x;
  const int lane = tid & 31, wv = tid >> 5;
  const int col  = lane & 15, half = lane >> 4;
  const int i0   = blockIdx.x * 16;
  {   // stage A tile (16 points x 128 feats) as b128 copies: 1 chunk per thread
    const int r = tid >> 4, c8 = tid & 15;
    *(uint4*)(abuf + r * 128 + c8 * 8) = *(const uint4*)(xb + (size_t)(i0 + r) * 128 + c8 * 8);
  }
  BFrag b1[4];
#pragma unroll
  for (int kt = 0; kt < 4; ++kt) {
    const unsigned int* p = w1ap + (wv * 4 + kt) * 256 + lane * 8;
#pragma unroll
    for (int v = 0; v < 8; ++v) b1[kt].u[v] = p[v];
  }
  __syncthreads();
  v8f c1 = {};
#pragma unroll
  for (int kt = 0; kt < 4; ++kt) {
    BFrag a;
    const unsigned short* rp = abuf + (lane & 15) * 128 + kt * 32 + half * 8;
#pragma unroll
    for (int v = 0; v < 8; ++v)
      a.u[v] = *(const unsigned int*)(rp + ((v & 3) * 2) + ((v >> 2) * 16));
    c1 = __builtin_amdgcn_wmma_f32_16x16x32_bf16(false, a.v, false, b1[kt].v,
                                                 (short)0, c1, false, false);
  }
#pragma unroll
  for (int r = 0; r < 8; ++r)
    selfc[(size_t)(i0 + r + 8 * half) * 128 + wv * 16 + col] = c1[r];
}

// ---------------- message MLP + max aggregation (bf16 WMMA, weights VGPR-resident) ----------------
// layer1 A = [neigh_x(128) | rel(4) | pad(28)] (K=160), self term added per column from selfc.
__global__ void __launch_bounds__(256) k_msg(const unsigned short* __restrict__ xb,
                                             const float* __restrict__ coords,
                                             const int* __restrict__ knn,
                                             const float* __restrict__ selfc,
                                             const unsigned int* __restrict__ w1p,
                                             const unsigned int* __restrict__ w2p,
                                             const float* __restrict__ mb1,
                                             const float* __restrict__ mb2,
                                             unsigned short* __restrict__ aggr) {
  __shared__ __align__(16) unsigned short msg[16 * 160];   // [neigh_x|rel|pad] bf16
  __shared__ __align__(16) unsigned short hbuf[16 * 128];  // relu(layer1) bf16
  const int tid  = threadIdx.x;
  const int lane = tid & 31, wv = tid >> 5;
  const int col  = lane & 15, half = lane >> 4;

  // zero the static pad columns (132..159) once per block
  for (int p = tid; p < 16 * 28; p += 256) {
    int r = p / 28, c = 132 + p % 28;
    msg[r * 160 + c] = 0;
  }

  // this wave's B column strip (cols 16*wv..16*wv+15), all K tiles, VGPR-resident
  BFrag b1[5], b2[4];
#pragma unroll
  for (int kt = 0; kt < 5; ++kt) {
    const unsigned int* p = w1p + (wv * 5 + kt) * 256 + lane * 8;
#pragma unroll
    for (int v = 0; v < 8; ++v) b1[kt].u[v] = p[v];
  }
#pragma unroll
  for (int kt = 0; kt < 4; ++kt) {
    const unsigned int* p = w2p + (wv * 4 + kt) * 256 + lane * 8;
#pragma unroll
    for (int v = 0; v < 8; ++v) b2[kt].u[v] = p[v];
  }
  const float bias1 = mb1[wv * 16 + col];
  const float bias2 = mb2[wv * 16 + col];

  for (int pi = 0; pi < PTS_PER_BLK; ++pi) {
    const int i = blockIdx.x * PTS_PER_BLK + pi;
    {   // neighbor-feature gather: one b128 chunk per thread (16 rows x 16 chunks)
      const int r = tid >> 4, c8 = tid & 15;
      const int j = knn[i * KNN + r];
      *(uint4*)(msg + r * 160 + c8 * 8) = *(const uint4*)(xb + (size_t)j * 128 + c8 * 8);
    }
    if (tid < 64) {   // rel coords (4 per row)
      const int r = tid >> 2, comp = tid & 3;
      const int j = knn[i * KNN + r];
      msg[r * 160 + 128 + comp] = f2bf(coords[(size_t)j * 4 + comp] -
                                       coords[(size_t)i * 4 + comp]);
    }
    __syncthreads();

    v8f c1 = {};
#pragma unroll
    for (int kt = 0; kt < 5; ++kt) {          // K = 160
      BFrag a;
      const unsigned short* rp = msg + (lane & 15) * 160 + kt * 32 + half * 8;
#pragma unroll
      for (int v = 0; v < 8; ++v)
        a.u[v] = *(const unsigned int*)(rp + ((v & 3) * 2) + ((v >> 2) * 16));
      c1 = __builtin_amdgcn_wmma_f32_16x16x32_bf16(false, a.v, false, b1[kt].v,
                                                   (short)0, c1, false, false);
    }
    const float base = selfc[(size_t)i * 128 + wv * 16 + col] + bias1; // same for all M rows
#pragma unroll
    for (int r = 0; r < 8; ++r) {             // + self + bias, relu -> hbuf (bf16)
      float h = c1[r] + base;
      h = h > 0.f ? h : 0.f;
      hbuf[(r + 8 * half) * 128 + wv * 16 + col] = f2bf(h);
    }
    __syncthreads();

    v8f c2 = {};
#pragma unroll
    for (int kt = 0; kt < 4; ++kt) {          // K = 128
      BFrag a;
      const unsigned short* rp = hbuf + (lane & 15) * 128 + kt * 32 + half * 8;
#pragma unroll
      for (int v = 0; v < 8; ++v)
        a.u[v] = *(const unsigned int*)(rp + ((v & 3) * 2) + ((v >> 2) * 16));
      c2 = __builtin_amdgcn_wmma_f32_16x16x32_bf16(false, a.v, false, b2[kt].v,
                                                   (short)0, c2, false, false);
    }
    // max over the 16 neighbor rows: 8 C-VGPRs cover M = r + 8*half, then fold halves
    float m = -3.4e38f;
#pragma unroll
    for (int r = 0; r < 8; ++r) {
      float cv = c2[r] + bias2;
      m = cv > m ? cv : m;
    }
    float o = __shfl_xor(m, 16, 32);
    m = o > m ? o : m;
    if (lane < 16) aggr[(size_t)i * 128 + wv * 16 + lane] = f2bf(m);
    __syncthreads();
  }
}

// ---------------- update MLP: [x|aggr] (N x 256) -> relu -> (N x 128) ----------------
__global__ void __launch_bounds__(256) k_upd(const unsigned short* __restrict__ xb,
                                             const unsigned short* __restrict__ aggr,
                                             const unsigned int* __restrict__ u1p,
                                             const unsigned int* __restrict__ u2p,
                                             const float* __restrict__ ub1,
                                             const float* __restrict__ ub2,
                                             float* __restrict__ out) {
  __shared__ __align__(16) unsigned short abuf[16 * 256];
  __shared__ __align__(16) unsigned short hbuf[16 * 128];
  const int tid  = threadIdx.x;
  const int lane = tid & 31, wv = tid >> 5;
  const int col  = lane & 15, half = lane >> 4;
  const int i0   = blockIdx.x * 16;

  {   // stage A tile (16 points x 256 feats) as b128 copies: 2 chunks per thread
    const int r = tid >> 4, c8 = tid & 15;
    *(uint4*)(abuf + r * 256 + c8 * 8)       = *(const uint4*)(xb   + (size_t)(i0 + r) * 128 + c8 * 8);
    *(uint4*)(abuf + r * 256 + 128 + c8 * 8) = *(const uint4*)(aggr + (size_t)(i0 + r) * 128 + c8 * 8);
  }
  BFrag b1[8], b2[4];
#pragma unroll
  for (int kt = 0; kt < 8; ++kt) {
    const unsigned int* p = u1p + (wv * 8 + kt) * 256 + lane * 8;
#pragma unroll
    for (int v = 0; v < 8; ++v) b1[kt].u[v] = p[v];
  }
#pragma unroll
  for (int kt = 0; kt < 4; ++kt) {
    const unsigned int* p = u2p + (wv * 4 + kt) * 256 + lane * 8;
#pragma unroll
    for (int v = 0; v < 8; ++v) b2[kt].u[v] = p[v];
  }
  __syncthreads();

  v8f c1 = {};
#pragma unroll
  for (int kt = 0; kt < 8; ++kt) {            // K = 256
    BFrag a;
    const unsigned short* rp = abuf + (lane & 15) * 256 + kt * 32 + half * 8;
#pragma unroll
    for (int v = 0; v < 8; ++v)
      a.u[v] = *(const unsigned int*)(rp + ((v & 3) * 2) + ((v >> 2) * 16));
    c1 = __builtin_amdgcn_wmma_f32_16x16x32_bf16(false, a.v, false, b1[kt].v,
                                                 (short)0, c1, false, false);
  }
  const float bias1 = ub1[wv * 16 + col];
#pragma unroll
  for (int r = 0; r < 8; ++r) {
    float h = c1[r] + bias1;
    h = h > 0.f ? h : 0.f;
    hbuf[(r + 8 * half) * 128 + wv * 16 + col] = f2bf(h);
  }
  __syncthreads();

  v8f c2 = {};
#pragma unroll
  for (int kt = 0; kt < 4; ++kt) {            // K = 128
    BFrag a;
    const unsigned short* rp = hbuf + (lane & 15) * 128 + kt * 32 + half * 8;
#pragma unroll
    for (int v = 0; v < 8; ++v)
      a.u[v] = *(const unsigned int*)(rp + ((v & 3) * 2) + ((v >> 2) * 16));
    c2 = __builtin_amdgcn_wmma_f32_16x16x32_bf16(false, a.v, false, b2[kt].v,
                                                 (short)0, c2, false, false);
  }
  const float bias2 = ub2[wv * 16 + col];
#pragma unroll
  for (int r = 0; r < 8; ++r)
    out[(size_t)(i0 + r + 8 * half) * 128 + wv * 16 + col] = c2[r] + bias2;
}

// ---------------- launcher ----------------
extern "C" void kernel_launch(void* const* d_in, const int* in_sizes, int n_in,
                              void* d_out, int out_size, void* d_ws, size_t ws_size,
                              hipStream_t stream) {
  (void)in_sizes; (void)n_in; (void)out_size; (void)ws_size;
  const float* x      = (const float*)d_in[0];
  const float* coords = (const float*)d_in[1];
  const float* mW1    = (const float*)d_in[2];
  const float* mb1    = (const float*)d_in[3];
  const float* mW2    = (const float*)d_in[4];
  const float* mb2    = (const float*)d_in[5];
  const float* uW1    = (const float*)d_in[6];
  const float* ub1    = (const float*)d_in[7];
  const float* uW2    = (const float*)d_in[8];
  const float* ub2    = (const float*)d_in[9];
  float* out = (float*)d_out;

  char* ws = (char*)d_ws;
  size_t off = 0;
  auto carve = [&](size_t bytes) -> void* {
    void* p = ws + off;
    off = (off + bytes + 255) & ~(size_t)255;
    return p;
  };
  unsigned short* xb    = (unsigned short*)carve((size_t)N_PTS * 128 * 2);
  int*            nbr   = (int*)carve((size_t)N_PTS * KNN * 4);
  unsigned short* aggr  = (unsigned short*)carve((size_t)N_PTS * 128 * 2);
  float*          selfc = (float*)carve((size_t)N_PTS * 128 * 4);
  unsigned int*   w1ap  = (unsigned int*)carve(32 * 256 * 4);   // self  weights: 8 nt x 4 kt
  unsigned int*   w1bp  = (unsigned int*)carve(40 * 256 * 4);   // neigh+rel:     8 nt x 5 kt
  unsigned int*   w2p   = (unsigned int*)carve(32 * 256 * 4);   // 8 x 4
  unsigned int*   u1p   = (unsigned int*)carve(64 * 256 * 4);   // 8 x 8
  unsigned int*   u2p   = (unsigned int*)carve(32 * 256 * 4);   // 8 x 4

  k_conv_x<<<(N_PTS * 128 / 4 + 255) / 256, 256, 0, stream>>>((const float4*)x, (uint2*)xb,
                                                              N_PTS * 128 / 4);
  k_pack_w<<<32, 256, 0, stream>>>(mW1, w1ap, 128, 4);              // rows 0..127 (self)
  k_pack_w<<<40, 256, 0, stream>>>(mW1 + 128 * 128, w1bp, 132, 5);  // rows 128..259, pad->160
  k_pack_w<<<32, 256, 0, stream>>>(mW2, w2p, 128, 4);
  k_pack_w<<<64, 256, 0, stream>>>(uW1, u1p, 256, 8);
  k_pack_w<<<32, 256, 0, stream>>>(uW2, u2p, 128, 4);
  k_knn<<<N_PTS / 8, 256, 0, stream>>>(coords, nbr);
  k_selfc<<<N_PTS / 16, 256, 0, stream>>>(xb, w1ap, selfc);
  k_msg<<<N_PTS / PTS_PER_BLK, 256, 0, stream>>>(xb, coords, nbr, selfc, w1bp, w2p,
                                                 mb1, mb2, aggr);
  k_upd<<<N_PTS / 16, 256, 0, stream>>>(xb, aggr, u1p, u2p, ub1, ub2, out);
}
